// CPMAntAttention_12618613916424
// MI455X (gfx1250) — compile-verified, matrix operand order checked by reading, not run on hardware
//
#include <hip/hip_runtime.h>
#include <hip/hip_bf16.h>

#define B_  2
#define S_  1024
#define D_  4096
#define H_  32
#define DH_ 128

typedef __attribute__((ext_vector_type(16))) __bf16 v16bf;
typedef __attribute__((ext_vector_type(8)))  float  v8f;
typedef __attribute__((ext_vector_type(4)))  unsigned int v4u;
typedef __attribute__((ext_vector_type(8)))  int v8i;
typedef __attribute__((ext_vector_type(4)))  int v4i;

__device__ __forceinline__ unsigned short f2bf(float f) {
    unsigned int u = __float_as_uint(f);
    u += 0x7FFFu + ((u >> 16) & 1u);          // round-to-nearest-even
    return (unsigned short)(u >> 16);
}

union FragBF {
    uint4 u[2];
    v16bf v;
};

__device__ __forceinline__ v8f wmma_bf16(v16bf a, v16bf b, v8f c) {
    return __builtin_amdgcn_wmma_f32_16x16x32_bf16(false, a, false, b, (short)0, c,
                                                   false, false);
}

__device__ __forceinline__ v8f v8f_zero() {
    v8f z = {0.f, 0.f, 0.f, 0.f, 0.f, 0.f, 0.f, 0.f};
    return z;
}

// ---------------------------------------------------------------------------
// TDM: async DMA of a 2D bf16 tile (tile_d0 contiguous elems x tile_d1 rows,
// row pitch stride0 elems) from global memory into LDS at lds_off.
// D# per CDNA5 ISA. Groups 2/3 zero (2-D tensor).
// ---------------------------------------------------------------------------
__device__ __forceinline__ void tdm_load_2d(unsigned lds_off,
                                            unsigned long long gaddr,
                                            unsigned tile_d0, unsigned tile_d1,
                                            unsigned long long stride0)
{
    v4u g0;
    g0[0] = 1u;                                        // count=1 (valid, user mode)
    g0[1] = lds_off;                                   // lds_addr (bytes)
    g0[2] = (unsigned)(gaddr & 0xFFFFFFFFu);           // global_addr[31:0]
    g0[3] = (unsigned)((gaddr >> 32) & 0x01FFFFFFu)    // global_addr[56:32]
          | (2u << 30);                                // type=2 ("image")
    v8i g1;
    g1[0] = (int)(1u << 16);                           // wg_mask=0, data_size=1 (2B)
    g1[1] = (int)((tile_d0 & 0xFFFFu) << 16);          // tensor_dim0[15:0] @ [63:48]
    g1[2] = (int)(((tile_d0 >> 16) & 0xFFFFu)          // tensor_dim0[31:16]
          | ((tile_d1 & 0xFFFFu) << 16));              // tensor_dim1[15:0]
    g1[3] = (int)(((tile_d1 >> 16) & 0xFFFFu)          // tensor_dim1[31:16]
          | ((tile_d0 & 0xFFFFu) << 16));              // tile_dim0
    g1[4] = (int)(tile_d1 & 0xFFFFu);                  // tile_dim1 ; tile_dim2=0 (2-D)
    g1[5] = (int)(unsigned)(stride0 & 0xFFFFFFFFu);    // tensor_dim0_stride[31:0]
    g1[6] = (int)((stride0 >> 32) & 0xFFFFu);          // stride0[47:32]; dim1_stride lo=0
    g1[7] = 0;
    v4i z4 = {0, 0, 0, 0};
#if __clang_major__ >= 23
    v8i z8 = {0, 0, 0, 0, 0, 0, 0, 0};
    __builtin_amdgcn_tensor_load_to_lds(g0, g1, z4, z4, z8, 0);
#else
    __builtin_amdgcn_tensor_load_to_lds(g0, g1, z4, z4, 0);
#endif
}

__device__ __forceinline__ unsigned lds_offset_of(const void* p) {
    // LDS aperture generic address: low 32 bits == LDS byte offset (ISA S10.2)
    return (unsigned)(unsigned long long)(uintptr_t)p;
}

// ---------------------------------------------------------------------------
// Generic tiled GEMM: C[M,N] = A[M,K] (f32, row-major) x B[K,N] (f32, row-major)
// Converted to bf16 in LDS, computed with v_wmma_f32_16x16x32_bf16.
// Block: 256 threads (8 waves), 64x64 output tile, K step 32.
// MODE 0: store f32 row-major (ld = N)
// MODE 1: store bf16 as (B,H,S,DH)   [Q / K]
// MODE 2: store bf16 as (B,H,DH,S)   [V transposed]
// ---------------------------------------------------------------------------
template <int MODE>
__global__ __launch_bounds__(256)
void gemm_wmma(const float* __restrict__ A, const float* __restrict__ Bm,
               void* __restrict__ Out, int M, int N, int K)
{
    __shared__ __align__(16) unsigned short Als[64 * 32];  // [m][k]
    __shared__ __align__(16) unsigned short Bls[64 * 32];  // [n][k] (transposed)

    const int tid  = threadIdx.x;
    const int lane = tid & 31;
    const int wave = tid >> 5;
    const int hi   = lane >> 4;
    const int l16  = lane & 15;

    const int n0 = blockIdx.x * 64;
    const int m0 = blockIdx.y * 64;

    const int m_off = (wave & 3) * 16;   // 4 row-tiles
    const int n_off = (wave >> 2) * 32;  // 2 col-tiles per wave

    v8f acc0 = v8f_zero();
    v8f acc1 = v8f_zero();

    const int arow = tid >> 2;           // 0..63
    const int acg  = (tid & 3) * 8;      // k-group of 8
    const int bk   = tid >> 3;           // 0..31 (k row of B)
    const int bng  = (tid & 7) * 8;      // n-group of 8

    for (int kk = 0; kk < K; kk += 32) {
        __syncthreads();
        // ---- stage A tile 64x32 (f32 -> bf16, row-major) ----
        {
            const float4* ap = (const float4*)(A + (size_t)(m0 + arow) * K + kk + acg);
            float4 a0 = ap[0], a1 = ap[1];
            __builtin_prefetch(ap + 8, 0, 1);
            uint4 pk;
            pk.x = (unsigned)f2bf(a0.x) | ((unsigned)f2bf(a0.y) << 16);
            pk.y = (unsigned)f2bf(a0.z) | ((unsigned)f2bf(a0.w) << 16);
            pk.z = (unsigned)f2bf(a1.x) | ((unsigned)f2bf(a1.y) << 16);
            pk.w = (unsigned)f2bf(a1.z) | ((unsigned)f2bf(a1.w) << 16);
            ((uint4*)Als)[arow * 4 + (tid & 3)] = pk;
        }
        // ---- stage B tile 32x64 transposed into [n][k] ----
        {
            const float4* bp = (const float4*)(Bm + (size_t)(kk + bk) * N + n0 + bng);
            float4 b0 = bp[0], b1 = bp[1];
            float bv[8] = {b0.x, b0.y, b0.z, b0.w, b1.x, b1.y, b1.z, b1.w};
            #pragma unroll
            for (int i = 0; i < 8; ++i)
                Bls[(size_t)(bng + i) * 32 + bk] = f2bf(bv[i]);
        }
        __syncthreads();

        // ---- fragments + WMMA ----
        FragBF af, bf0, bf1;
        const int arw = m_off + l16;
        af.u[0]  = *(const uint4*)&Als[arw * 32 + hi * 8];
        af.u[1]  = *(const uint4*)&Als[arw * 32 + 16 + hi * 8];
        const int br0 = n_off + l16;
        bf0.u[0] = *(const uint4*)&Bls[br0 * 32 + hi * 8];
        bf0.u[1] = *(const uint4*)&Bls[br0 * 32 + 16 + hi * 8];
        const int br1 = n_off + 16 + l16;
        bf1.u[0] = *(const uint4*)&Bls[br1 * 32 + hi * 8];
        bf1.u[1] = *(const uint4*)&Bls[br1 * 32 + 16 + hi * 8];

        acc0 = wmma_bf16(af.v, bf0.v, acc0);
        acc1 = wmma_bf16(af.v, bf1.v, acc1);
    }

    // ---- store: element (m,n): lane = n%16 + 16*(m%16 >= 8), vgpr = m%8 ----
    #pragma unroll
    for (int r = 0; r < 8; ++r) {
        const int m = m0 + m_off + r + 8 * hi;
        #pragma unroll
        for (int s = 0; s < 2; ++s) {
            const int n = n0 + n_off + s * 16 + l16;
            const float val = (s == 0) ? acc0[r] : acc1[r];
            if (MODE == 0) {
                ((float*)Out)[(size_t)m * N + n] = val;
            } else {
                const int b  = m >> 10, sp = m & (S_ - 1);
                const int h  = n >> 7,  dh = n & (DH_ - 1);
                if (MODE == 1)
                    ((unsigned short*)Out)[(((size_t)b * H_ + h) * S_ + sp) * DH_ + dh] = f2bf(val);
                else
                    ((unsigned short*)Out)[(((size_t)b * H_ + h) * DH_ + dh) * S_ + sp] = f2bf(val);
            }
        }
    }
}

// ---------------------------------------------------------------------------
// Flash attention: block = 128 threads (4 waves); wave owns a 16-row Q tile.
// K/V tiles staged into LDS by TDM (tensor_load_to_lds), double-buffered so
// DMA of key-block i+1 overlaps WMMA compute of block i.
// Q,K: (B,H,S,DH) bf16.  Vt: (B,H,DH,S) bf16.  ctx out: (B,S,H*DH) f32.
// ---------------------------------------------------------------------------
__global__ __launch_bounds__(128)
void flash_attn(const unsigned short* __restrict__ Q,
                const unsigned short* __restrict__ Km,
                const unsigned short* __restrict__ Vt,
                const float* __restrict__ bias,
                const unsigned char* __restrict__ mask,
                float* __restrict__ ctx)
{
    __shared__ __align__(16) unsigned short Kls[2][32 * DH_];  // [key][dh] 8KB x2
    __shared__ __align__(16) unsigned short Vls[2][DH_ * 32];  // [dh][key] 8KB x2
    __shared__ __align__(16) unsigned short Pls[4][16 * 32];   // per-wave P staging

    const int tid  = threadIdx.x;
    const int lane = tid & 31;
    const int wave = tid >> 5;
    const int hi   = lane >> 4;
    const int l16  = lane & 15;

    const int h  = blockIdx.y;
    const int b  = blockIdx.z;
    const int q0 = blockIdx.x * 64 + wave * 16;

    const size_t bh = (size_t)b * H_ + h;
    const unsigned short* Kbase = Km + bh * S_ * DH_;  // rows: DH_-contiguous
    const unsigned short* Vbase = Vt + bh * DH_ * S_;  // rows: S_-contiguous

    // Q fragments: 4 chunks of K=32 over DH=128 (rows dh-contiguous in memory)
    FragBF qf[4];
    {
        const unsigned short* qp = Q + (bh * S_ + (q0 + l16)) * DH_;
        #pragma unroll
        for (int j = 0; j < 4; ++j) {
            qf[j].u[0] = *(const uint4*)&qp[j * 32 + hi * 8];
            qf[j].u[1] = *(const uint4*)&qp[j * 32 + 16 + hi * 8];
        }
    }

    v8f O[8];
    #pragma unroll
    for (int t = 0; t < 8; ++t) O[t] = v8f_zero();
    float mrow[8], lrow[8];
    #pragma unroll
    for (int r = 0; r < 8; ++r) { mrow[r] = -3.0e38f; lrow[r] = 0.f; }

    const float scale = 0.08838834764831845f;  // 1/sqrt(128)

    // prologue: DMA first K/V tiles into buffer 0
    if (wave == 0) {
        tdm_load_2d(lds_offset_of(&Kls[0][0]),
                    (unsigned long long)(uintptr_t)(Kbase), DH_, 32, DH_);
        tdm_load_2d(lds_offset_of(&Vls[0][0]),
                    (unsigned long long)(uintptr_t)(Vbase), 32, DH_, S_);
        __builtin_amdgcn_s_wait_tensorcnt(0);
    }
    __syncthreads();

    int buf = 0;
    for (int k0 = 0; k0 < S_; k0 += 32) {
        // kick off DMA for the next key block into the other buffer
        if (wave == 0 && (k0 + 32) < S_) {
            tdm_load_2d(lds_offset_of(&Kls[buf ^ 1][0]),
                        (unsigned long long)(uintptr_t)(Kbase + (size_t)(k0 + 32) * DH_),
                        DH_, 32, DH_);
            tdm_load_2d(lds_offset_of(&Vls[buf ^ 1][0]),
                        (unsigned long long)(uintptr_t)(Vbase + (k0 + 32)),
                        32, DH_, S_);
        }

        float sc[2][8];
        #pragma unroll
        for (int sub = 0; sub < 2; ++sub) {
            v8f acc = v8f_zero();
            const int krl = sub * 16 + l16;                 // key row within tile
            const unsigned short* kp = &Kls[buf][krl * DH_];
            #pragma unroll
            for (int j = 0; j < 4; ++j) {
                FragBF kf;
                kf.u[0] = *(const uint4*)&kp[j * 32 + hi * 8];
                kf.u[1] = *(const uint4*)&kp[j * 32 + 16 + hi * 8];
                acc = wmma_bf16(qf[j].v, kf.v, acc);
            }
            const int krow = k0 + krl;
            #pragma unroll
            for (int r = 0; r < 8; ++r) {
                const int q = q0 + r + 8 * hi;
                float v = acc[r] * scale + bias[(bh * S_ + q) * S_ + krow];
                if (!mask[((size_t)b * S_ + q) * S_ + krow]) v = -3.0e38f;
                sc[sub][r] = v;
            }
        }

        // online softmax per row (row stats live replicated across 16-lane half)
        #pragma unroll
        for (int r = 0; r < 8; ++r) {
            float mx = fmaxf(sc[0][r], sc[1][r]);
            #pragma unroll
            for (int off = 8; off >= 1; off >>= 1)
                mx = fmaxf(mx, __shfl_xor(mx, off, 32));
            const float mnew = fmaxf(mrow[r], mx);
            const float corr = __expf(mrow[r] - mnew);
            const float p0 = (sc[0][r] <= -1.0e37f) ? 0.f : __expf(sc[0][r] - mnew);
            const float p1 = (sc[1][r] <= -1.0e37f) ? 0.f : __expf(sc[1][r] - mnew);
            float rs = p0 + p1;
            #pragma unroll
            for (int off = 8; off >= 1; off >>= 1)
                rs += __shfl_xor(rs, off, 32);
            lrow[r] = lrow[r] * corr + rs;
            mrow[r] = mnew;
            #pragma unroll
            for (int t = 0; t < 8; ++t) O[t][r] *= corr;
            const int prow = r + 8 * hi;
            Pls[wave][prow * 32 + l16]      = f2bf(p0);
            Pls[wave][prow * 32 + 16 + l16] = f2bf(p1);
        }

        // per-wave LDS round trip (D-layout -> A-layout); LDS is in-order per wave
        asm volatile("s_wait_dscnt 0" ::: "memory");

        FragBF pf;
        pf.u[0] = *(const uint4*)&Pls[wave][l16 * 32 + hi * 8];
        pf.u[1] = *(const uint4*)&Pls[wave][l16 * 32 + 16 + hi * 8];

        // O += P(16x32) x V(32xDH), Vls rows are key-contiguous
        #pragma unroll
        for (int t = 0; t < 8; ++t) {
            FragBF vf;
            const unsigned short* vp = &Vls[buf][(t * 16 + l16) * 32];
            vf.u[0] = *(const uint4*)&vp[hi * 8];
            vf.u[1] = *(const uint4*)&vp[16 + hi * 8];
            O[t] = wmma_bf16(pf.v, vf.v, O[t]);
        }

        // publish next buffer: DMA must be done before anyone reads it
        if (wave == 0) __builtin_amdgcn_s_wait_tensorcnt(0);
        __syncthreads();
        buf ^= 1;
    }

    // normalize and store ctx (B,S,H*DH) f32
    #pragma unroll
    for (int t = 0; t < 8; ++t) {
        #pragma unroll
        for (int r = 0; r < 8; ++r) {
            const int q = q0 + r + 8 * hi;
            const float l = lrow[r];
            const float val = (l > 0.f) ? O[t][r] / l : 0.f;
            ctx[((size_t)b * S_ + q) * (H_ * DH_) + h * DH_ + t * 16 + l16] = val;
        }
    }
}

// ---------------------------------------------------------------------------
extern "C" void kernel_launch(void* const* d_in, const int* in_sizes, int n_in,
                              void* d_out, int out_size, void* d_ws, size_t ws_size,
                              hipStream_t stream)
{
    (void)in_sizes; (void)n_in; (void)out_size; (void)ws_size;

    const float*         hidden_q  = (const float*)d_in[0];
    const float*         hidden_kv = (const float*)d_in[1];
    const unsigned char* mask      = (const unsigned char*)d_in[2];
    const float*         bias      = (const float*)d_in[3];
    const float*         Wq        = (const float*)d_in[4];
    const float*         Wk        = (const float*)d_in[5];
    const float*         Wv        = (const float*)d_in[6];
    const float*         Wo        = (const float*)d_in[7];

    const size_t QKV_ELEMS = (size_t)B_ * H_ * S_ * DH_;  // 8,388,608
    unsigned short* Qb = (unsigned short*)d_ws;
    unsigned short* Kb = Qb + QKV_ELEMS;
    unsigned short* Vt = Kb + QKV_ELEMS;
    float*          ctx = (float*)(Vt + QKV_ELEMS);

    const dim3 gblk(256);
    const dim3 ggrid(D_ / 64, (B_ * S_) / 64);  // (64, 32)

    gemm_wmma<1><<<ggrid, gblk, 0, stream>>>(hidden_q,  Wq, Qb, B_ * S_, H_ * DH_, D_);
    gemm_wmma<1><<<ggrid, gblk, 0, stream>>>(hidden_kv, Wk, Kb, B_ * S_, H_ * DH_, D_);
    gemm_wmma<2><<<ggrid, gblk, 0, stream>>>(hidden_kv, Wv, Vt, B_ * S_, H_ * DH_, D_);

    const dim3 agrid(S_ / 64, H_, B_);
    flash_attn<<<agrid, dim3(128), 0, stream>>>(Qb, Kb, Vt, bias, mask, ctx);

    gemm_wmma<0><<<ggrid, gblk, 0, stream>>>(ctx, Wo, d_out, B_ * S_, D_, H_ * DH_);
}
